// CachedConvNet_3908420239787
// MI455X (gfx1250) — matrix operand order repaired and checked by reading, not compiled, not run on hardware
//
#include <hip/hip_runtime.h>

typedef _Float16 h16;
typedef __attribute__((ext_vector_type(16))) _Float16 v16h;
typedef __attribute__((ext_vector_type(8)))  _Float16 v8h;
typedef __attribute__((ext_vector_type(8)))  float    v8f;

#define TILE  256            // output time samples per block
#define HALO  128            // activation rows [HALO, HALO+TILE) are the tile; halo in [2,HALO)
#define ROWS  (TILE + HALO)  // 384
#define PITCH 72             // halves per row (64 ch + 8 pad) = 144 B -> conflict-free b128 loads
#define NTHREADS 256
#define NWAVES   8

static constexpr int L = 6, C = 64, KSZ = 3, TLEN = 65536, BATCH = 8;
static constexpr int OFFS[7] = {0, 2, 6, 14, 30, 62, 126};  // cumsum of (K-1)*dil
// pre-swizzled weight image in d_ws: [layer][mtile][frag(conv,k,kstep)][lane][16 halves]
#define FRAGS_PER_MT 12                       // 2 convs * 3 taps * 2 ksteps
#define WS_HALVES (L * 4 * FRAGS_PER_MT * 512)  // 147456 halves = 288 KB

__device__ __forceinline__ v16h ld_frag16(const h16* p) {
  v8h lo = *(const v8h*)(p);
  v8h hi = *(const v8h*)(p + 8);
  return __builtin_shufflevector(lo, hi, 0,1,2,3,4,5,6,7,8,9,10,11,12,13,14,15);
}

// ---------------- prep: swizzle f32 weights into WMMA A-fragment order (f16) ----------------
// One element per thread; runs once per launch, output reused by all 2048 main blocks.
__global__ __launch_bounds__(NTHREADS)
void weight_swizzle_kernel(const float* __restrict__ Wf, const float* __restrict__ Wg,
                           h16* __restrict__ ws) {
  int idx = blockIdx.x * NTHREADS + threadIdx.x;        // [0, WS_HALVES)
  int e    = idx & 15;
  int ln   = (idx >> 4) & 31;
  int rest = idx >> 9;                                  // ((i*4+mt)*12 + (cv*3+kk)*2 + ks)
  int ks   = rest & 1;
  int q    = rest >> 1;
  int ckk  = q % 6;                                     // cv*3 + kk
  int cv   = ckk / 3;
  int kk   = ckk % 3;
  int q2   = q / 6;
  int mt   = q2 & 3;
  int i    = q2 >> 2;

  int cout = mt * 16 + (ln & 15);
  int lhi  = ln >> 4;
  int kidx = lhi * 16 + e;                              // contiguous k order: k = 16*half + e (matches B)
  int cin  = ks * 32 + kidx;
  const float* W = cv ? Wg : Wf;
  ws[idx] = (h16)W[((i * C + cout) * C + cin) * KSZ + kk];
}

// ---------------- main: fully fused 6-layer gated residual stack ----------------
__global__ __launch_bounds__(NTHREADS, 1)
void wavenet_fused_kernel(const float* __restrict__ x,   const float* __restrict__ ctx,
                          const h16*   __restrict__ ws,  const float* __restrict__ bf,
                          const float* __restrict__ bg,
                          float* __restrict__ out_x,     float* __restrict__ out_ctx)
{
  __shared__ h16 act[2][ROWS * PITCH];         // 2 * 384*72*2B = 110592 B total LDS

  const int tid  = threadIdx.x;
  const int lane = tid & 31;
  const int wave = tid >> 5;
  const int hi   = lane >> 4;    // half-wave id (0/1)
  const int nc   = lane & 15;    // column (time) id within 16-wide tile
  const int mt   = wave & 3;     // fixed c_out tile per wave -> A frags loop-invariant
  const int wslc = wave >> 2;    // 2 waves share each mtile, alternate n-tiles

  const int tilesPerBatch = TLEN / TILE;
  const int b    = blockIdx.x / tilesPerBatch;
  const int tile = blockIdx.x % tilesPerBatch;
  const long t0  = (long)tile * TILE;
  const bool first = (tile == 0);
  const bool last  = (tile == tilesPerBatch - 1);

  // ---- stage 0: load x tile (and interior halo) into act[0] as f16 ----
  for (int idx = tid; idx < C * TILE; idx += NTHREADS) {
    int c = idx >> 8;                 // TILE == 256
    int t = idx & (TILE - 1);
    float v = x[((long)(b * C + c) << 16) + t0 + t];
    act[0][(HALO + t) * PITCH + c] = (h16)v;
  }
  if (!first) {
    for (int idx = tid; idx < C * 126; idx += NTHREADS) {
      int c = idx / 126;
      int j = idx % 126;
      float v = x[((long)(b * C + c) << 16) + (t0 - 126) + j];
      act[0][(2 + j) * PITCH + c] = (h16)v;
    }
  }
  __syncthreads();

  int cur = 0;
#pragma unroll
  for (int i = 0; i < L; i++) {
    const int d   = 1 << i;
    const int buf = 2 * d;
    const int lb  = first ? HALO : (2 + OFFS[i + 1]);  // lowest output row this layer

    // ---- load this wave's 12 A-fragments straight into registers (coalesced b128) ----
    const h16* wbase = ws + (size_t)((i * 4 + mt) * FRAGS_PER_MT) * 512;
    v16h afrag[FRAGS_PER_MT];
#pragma unroll
    for (int fr = 0; fr < FRAGS_PER_MT; fr++)
      afrag[fr] = ld_frag16(wbase + fr * 512 + lane * 16);

    // prefetch next layer's fragment block into cache while we compute this layer
    if (i + 1 < L) {
      const h16* nb = ws + (size_t)(((i + 1) * 4 + mt) * FRAGS_PER_MT) * 512;
#pragma unroll
      for (int pfo = 0; pfo < 3; pfo++)
        __builtin_prefetch((const void*)(nb + lane * 192 + pfo * 64), 0, 3);
    }

    // per-lane biases for this wave's 8 output channels (contiguous -> vector loads)
    float bfr[8], bgr[8];
#pragma unroll
    for (int v = 0; v < 8; v++) {
      bfr[v] = bf[i * C + mt * 16 + hi * 8 + v];
      bgr[v] = bg[i * C + mt * 16 + hi * 8 + v];
    }

    // ---- first tile: inject cached context as this layer's input halo ----
    if (first) {
      for (int idx = tid; idx < C * buf; idx += NTHREADS) {
        int c = idx / buf;
        int j = idx % buf;
        act[cur][(HALO - buf + j) * PITCH + c] =
            (h16)ctx[(b * C + c) * 126 + OFFS[i] + j];
      }
    }
    // ---- last tile: emit updated cache (tail of this layer's input) ----
    if (last) {
      for (int idx = tid; idx < C * buf; idx += NTHREADS) {
        int c = idx / buf;
        int j = idx % buf;
        out_ctx[(b * C + c) * 126 + OFFS[i] + j] =
            (float)act[cur][(HALO + TILE - buf + j) * PITCH + c];
      }
    }
    __syncthreads();

    // ---- WMMA compute: read act[cur], write act[cur^1] rows [lb, ROWS) ----
    const h16* src = act[cur];
    h16*       dst = act[cur ^ 1];
    const int ntiles = (ROWS - lb + 15) >> 4;
    for (int nt = wslc; nt < ntiles; nt += 2) {
      int p0 = lb + (nt << 4);
      if (p0 > ROWS - 16) p0 = ROWS - 16;           // clamp (duplicate writes are identical)

      v8f facc = {};
      v8f gacc = {};
#pragma unroll
      for (int kk = 0; kk < 3; kk++) {
        const int rbase = p0 + nc - (2 - kk) * d;   // input time row for this tap
#pragma unroll
        for (int ks = 0; ks < 2; ks++) {
          // B fragment: B[K = ks*32 + hi*16 + e][N = nc] = act[row][c_in], contiguous halves
          v16h bfrag = ld_frag16(src + rbase * PITCH + ks * 32 + hi * 16);
          facc = __builtin_amdgcn_wmma_f32_16x16x32_f16(
              false, afrag[(0 * 3 + kk) * 2 + ks], false, bfrag, (short)0, facc, false, false);
          gacc = __builtin_amdgcn_wmma_f32_16x16x32_f16(
              false, afrag[(1 * 3 + kk) * 2 + ks], false, bfrag, (short)0, gacc, false, false);
        }
      }

      // epilogue: bias + tanh*sigmoid + residual (vector b128 LDS load/store)
      const int poff = (p0 + nc) * PITCH + mt * 16 + hi * 8;   // 8 contiguous channels
      v8h res8 = *(const v8h*)(src + poff);
      v8h y8;
#pragma unroll
      for (int v = 0; v < 8; v++) {
        float fv = facc[v] + bfr[v];
        float gv = gacc[v] + bgr[v];
        float th = 1.0f - 2.0f / (__expf(2.0f * fv) + 1.0f);   // tanh
        float sg = 1.0f / (1.0f + __expf(-gv));                // sigmoid
        y8[v] = (h16)((float)res8[v] + th * sg);
      }
      *(v8h*)(dst + poff) = y8;
    }
    __syncthreads();
    cur ^= 1;
  }

  // ---- write final x tile as f32 ----
  for (int idx = tid; idx < C * TILE; idx += NTHREADS) {
    int c = idx >> 8;
    int t = idx & (TILE - 1);
    out_x[((long)(b * C + c) << 16) + t0 + t] =
        (float)act[cur][(HALO + t) * PITCH + c];
  }
}

extern "C" void kernel_launch(void* const* d_in, const int* in_sizes, int n_in,
                              void* d_out, int out_size, void* d_ws, size_t ws_size,
                              hipStream_t stream) {
  const float* x   = (const float*)d_in[0];
  const float* ctx = (const float*)d_in[1];
  const float* Wf  = (const float*)d_in[2];
  const float* bf  = (const float*)d_in[3];
  const float* Wg  = (const float*)d_in[4];
  const float* bg  = (const float*)d_in[5];

  float* out_x   = (float*)d_out;
  float* out_ctx = out_x + (size_t)BATCH * C * TLEN;   // tuple (x, ctx) flattened
  h16*   ws      = (h16*)d_ws;                         // 288 KB pre-swizzled f16 weights

  // 1) one-time (per launch) weight swizzle: f32 -> f16 WMMA A-fragment image
  weight_swizzle_kernel<<<WS_HALVES / NTHREADS, NTHREADS, 0, stream>>>(Wf, Wg, ws);

  // 2) fused 6-layer stack, one block per (batch, 256-sample tile)
  dim3 grid(BATCH * (TLEN / TILE));                    // 2048 blocks
  wavenet_fused_kernel<<<grid, NTHREADS, 0, stream>>>(x, ctx, ws, bf, bg, out_x, out_ctx);
}
